// InICENODEStateFixedPoint_79044578116100
// MI455X (gfx1250) — matrix operand order repaired and checked by reading, not compiled, not run on hardware
//
#include <hip/hip_runtime.h>
#include <hip/hip_bf16.h>

// ---------------------------------------------------------------------------
// Problem constants (from reference)
// ---------------------------------------------------------------------------
#define S_DIM  512
#define H_DIM  1024
#define O_DIM  256
#define BATCH  512
#define NSTEPS 24
#define TRIALF 0.1f
#define EPSF   1e-12f

typedef __attribute__((ext_vector_type(16))) _Float16 v16h;
typedef __attribute__((ext_vector_type(8)))  _Float16 h8;
typedef __attribute__((ext_vector_type(8)))  float    v8f;

union V16U { v16h v; h8 h[2]; };

// ---------------------------------------------------------------------------
// WMMA GEMM, all-f16 inputs, f32 accumulate, double-buffered K pipeline:
//   C[M,N] = act( A[M,K] * B + bias ),  optionally masked by ReLU'(maskAct)
//   A: f16 row-major (lda = K)
//   W: f16; wtrans=1 -> B[k][n] = W[n*K+k] (forward, W^T)
//            wtrans=0 -> B[k][n] = W[k*N+n] (backward)
//   C: f32 out; C16: optional f16 mirror (feeds the next GEMM)
// Tile 64x64 per WG (4 waves, 128 thr), K stepped by 32. Next tile's global
// loads are issued into registers before the current tile's WMMAs, so L2
// latency overlaps matrix work; LDS ping-pongs between two buffers.
// LDS rows padded to 40 halves (80B) so every fragment is two ds_load_b128.
// Requires M%64==0, N%64==0, K%32==0 (true for all 9 shapes here).
// ---------------------------------------------------------------------------
#define TM 64
#define TN 64
#define TK 32
#define GEMM_THREADS 128
#define AS_LD 40   // halves per A row   (80B, 16B-aligned)
#define BT_LD 40   // halves per BsT row (80B, 16B-aligned)

__global__ __launch_bounds__(GEMM_THREADS)
void k_gemm_wmma(const _Float16* __restrict__ A,
                 const _Float16* __restrict__ W, int wtrans,
                 const float* __restrict__ bias,
                 const float* __restrict__ maskAct,
                 float* __restrict__ C,
                 _Float16* __restrict__ C16,
                 int M, int N, int K, int doRelu)
{
    __shared__ _Float16 As [2][TM * AS_LD];   // As[buf][m][k]
    __shared__ _Float16 BsT[2][TN * BT_LD];   // BsT[buf][n][k]

    const int tid  = threadIdx.x;
    const int wave = tid >> 5;        // 0..3 -> 16-row strip of the tile
    const int lane = tid & 31;
    const int lh   = lane & 15;       // row-in-strip (A/D) or N (B/D)
    const int kh   = lane >> 4;       // lane-half selector
    const int m0   = blockIdx.y * TM;
    const int n0   = blockIdx.x * TN;

    // fixed per-thread staging coordinates (2 h8 chunks each for A and B)
    const int f0 = tid * 8;                   // flat half-index, pass 0
    const int f1 = (tid + GEMM_THREADS) * 8;  // flat half-index, pass 1
    const int ar0 = f0 >> 5, ac0 = f0 & 31;   // A: row 0..63, col 0/8/16/24
    const int ar1 = f1 >> 5, ac1 = f1 & 31;

    v8f c[4];
    c[0] = v8f{}; c[1] = v8f{}; c[2] = v8f{}; c[3] = v8f{};

    h8 ra0, ra1, rb0, rb1;   // register-staged next tile

    auto loadTiles = [&](int k0) {
        ra0 = *(const h8*)&A[(m0 + ar0) * K + k0 + ac0];
        ra1 = *(const h8*)&A[(m0 + ar1) * K + k0 + ac1];
        if (wtrans) {          // W[n][k]: contiguous along k
            rb0 = *(const h8*)&W[(n0 + (f0 >> 5)) * K + k0 + (f0 & 31)];
            rb1 = *(const h8*)&W[(n0 + (f1 >> 5)) * K + k0 + (f1 & 31)];
        } else {               // W[k][n]: contiguous along n
            rb0 = *(const h8*)&W[(k0 + (f0 >> 6)) * N + n0 + (f0 & 63)];
            rb1 = *(const h8*)&W[(k0 + (f1 >> 6)) * N + n0 + (f1 & 63)];
        }
    };
    auto storeTiles = [&](int buf) {
        *(h8*)&As[buf][ar0 * AS_LD + ac0] = ra0;
        *(h8*)&As[buf][ar1 * AS_LD + ac1] = ra1;
        if (wtrans) {
            *(h8*)&BsT[buf][(f0 >> 5) * BT_LD + (f0 & 31)] = rb0;
            *(h8*)&BsT[buf][(f1 >> 5) * BT_LD + (f1 & 31)] = rb1;
        } else {               // scatter 8 halves along n into BsT[n][k]
            const int kk0 = f0 >> 6, nn0 = f0 & 63;
            const int kk1 = f1 >> 6, nn1 = f1 & 63;
            #pragma unroll
            for (int j = 0; j < 8; ++j) BsT[buf][(nn0 + j) * BT_LD + kk0] = rb0[j];
            #pragma unroll
            for (int j = 0; j < 8; ++j) BsT[buf][(nn1 + j) * BT_LD + kk1] = rb1[j];
        }
    };
    auto compute = [&](int buf) {
        V16U a;
        const int rowA = (wave * 16 + lh) * AS_LD;
        a.h[0] = *(const h8*)&As[buf][rowA + 8 * kh];
        a.h[1] = *(const h8*)&As[buf][rowA + 16 + 8 * kh];
        #pragma unroll
        for (int nt = 0; nt < 4; ++nt) {
            V16U b;
            const int rowB = (nt * 16 + lh) * BT_LD;
            b.h[0] = *(const h8*)&BsT[buf][rowB + 8 * kh];
            b.h[1] = *(const h8*)&BsT[buf][rowB + 16 + 8 * kh];
            c[nt] = __builtin_amdgcn_wmma_f32_16x16x32_f16(
                        false, a.v, false, b.v, (short)0, c[nt], false, false);
        }
    };

    // ---- software-pipelined K loop ----------------------------------------
    const int nk = K / TK;
    loadTiles(0);
    storeTiles(0);
    __syncthreads();
    for (int i = 0; i < nk; ++i) {
        if (i + 1 < nk) loadTiles((i + 1) * TK);   // overlap with WMMAs below
        compute(i & 1);
        if (i + 1 < nk) {
            __syncthreads();                       // all waves done with buf[(i+1)&1]
            storeTiles((i + 1) & 1);
            __syncthreads();                       // stores visible
        }
    }

    // ---- store: D layout M = v + 8*kh (within strip), N = lh --------------
    #pragma unroll
    for (int nt = 0; nt < 4; ++nt) {
        #pragma unroll
        for (int v = 0; v < 8; ++v) {
            int m = m0 + wave * 16 + v + 8 * kh;
            int n = n0 + nt * 16 + lh;
            float val = c[nt][v];
            if (bias)   val += bias[n];
            if (doRelu) val = val > 0.f ? val : 0.f;
            if (maskAct) val = (maskAct[m * N + n] > 0.f) ? val : 0.f;
            C[m * N + n] = val;
            if (C16) C16[m * N + n] = (_Float16)val;
        }
    }
}

// ---------------------------------------------------------------------------
// Per-sample block reduction helper (blockDim.x == 256)
// ---------------------------------------------------------------------------
__device__ inline void block_reduce2(float& a, float& b)
{
    __shared__ float r0[256], r1[256];
    int t = threadIdx.x;
    r0[t] = a; r1[t] = b;
    __syncthreads();
    for (int s = 128; s > 0; s >>= 1) {
        if (t < s) { r0[t] += r0[t + s]; r1[t] += r1[t + s]; }
        __syncthreads();
    }
    a = r0[0]; b = r1[0];
    __syncthreads();
}

// loss + (optional) dO (f32 + f16 mirror): one block (256 thr == O_DIM)/sample
__global__ __launch_bounds__(256)
void k_loss(const float* __restrict__ O, const float* __restrict__ Tobs,
            const int* __restrict__ Mask, float* __restrict__ F,
            float* __restrict__ dO, _Float16* __restrict__ dO16)
{
    int s = blockIdx.x, i = threadIdx.x;
    int idx = s * O_DIM + i;
    float m = (Mask[idx] != 0) ? 1.0f : 0.0f;
    float r = (Tobs[idx] - O[idx]) * m;
    float rsq = r * r, mm = m;
    block_reduce2(rsq, mm);
    float cnt = fmaxf(mm, 1.0f);
    if (i == 0) F[s] = rsq / cnt;
    if (dO) {
        float g = -2.0f * r / cnt;
        dO[idx]   = g;
        dO16[idx] = (_Float16)g;
    }
}

// direction fix (restart if not descent) + trial point (f16): block per sample
__global__ __launch_bounds__(256)
void k_dirfix_trial(const float* __restrict__ G, float* __restrict__ D,
                    const float* __restrict__ Y, _Float16* __restrict__ YTh,
                    float* __restrict__ GD)
{
    int s = blockIdx.x;
    const float* g = G + s * S_DIM;
    float*       d = D + s * S_DIM;
    const float* y = Y + s * S_DIM;
    _Float16*   yt = YTh + s * S_DIM;
    float gd = 0.f, gg = 0.f;
    for (int i = threadIdx.x; i < S_DIM; i += 256) {
        float gi = g[i];
        gd += gi * d[i];
        gg += gi * gi;
    }
    block_reduce2(gd, gg);
    bool descent = gd < 0.0f;
    for (int i = threadIdx.x; i < S_DIM; i += 256) {
        float dv = descent ? d[i] : -g[i];
        d[i]  = dv;
        yt[i] = (_Float16)(y[i] + TRIALF * dv);
    }
    if (threadIdx.x == 0) GD[s] = descent ? gd : -gg;
}

// quadratic-interpolation alpha + y_new (f32 + f16): block per sample
__global__ __launch_bounds__(256)
void k_alpha_step(const float* __restrict__ Y, const float* __restrict__ D,
                  const float* __restrict__ FT, const float* __restrict__ FC,
                  const float* __restrict__ GD, float* __restrict__ YN,
                  _Float16* __restrict__ YNh)
{
    int s = blockIdx.x;
    float ft = FT[s], fc = FC[s], gd = GD[s];
    float denom = 2.0f * (ft - fc - gd * TRIALF);
    float alpha = (denom > EPSF) ? (-gd * TRIALF * TRIALF / denom) : TRIALF;
    alpha = fminf(fmaxf(alpha, 0.0f), 10.0f);
    for (int i = threadIdx.x; i < S_DIM; i += 256) {
        int idx = s * S_DIM + i;
        float v = Y[idx] + alpha * D[idx];
        YN[idx]  = v;
        YNh[idx] = (_Float16)v;
    }
}

// PR+ beta, new direction, best-so-far, carry update: block per sample
__global__ __launch_bounds__(256)
void k_finalize(const float* __restrict__ GN, float* __restrict__ G,
                float* __restrict__ D, const float* __restrict__ YN,
                float* __restrict__ Y, const float* __restrict__ FN,
                float* __restrict__ FC, float* __restrict__ YB,
                float* __restrict__ FB)
{
    int s = blockIdx.x;
    const float* gn = GN + s * S_DIM;
    float*       g  = G  + s * S_DIM;
    float*       d  = D  + s * S_DIM;
    const float* yn = YN + s * S_DIM;
    float*       y  = Y  + s * S_DIM;
    float*       yb = YB + s * S_DIM;
    float num = 0.f, den = 0.f;
    for (int i = threadIdx.x; i < S_DIM; i += 256) {
        float gni = gn[i], gi = g[i];
        num += gni * (gni - gi);
        den += gi * gi;
    }
    block_reduce2(num, den);
    float beta = fmaxf(0.0f, num / fmaxf(den, EPSF));
    float fn = FN[s];
    bool better = fn < FB[s];
    for (int i = threadIdx.x; i < S_DIM; i += 256) {
        float gni = gn[i];
        float dn  = -gni + beta * d[i];
        float yv  = yn[i];
        if (better) yb[i] = yv;
        y[i] = yv; g[i] = gni; d[i] = dn;
    }
    if (threadIdx.x == 0) {
        if (better) FB[s] = fn;
        FC[s] = fn;
    }
}

// init: d = -g, y_best = y, f_best = f0
__global__ __launch_bounds__(256)
void k_init(const float* __restrict__ G, float* __restrict__ D,
            const float* __restrict__ Y, float* __restrict__ YB,
            const float* __restrict__ FC, float* __restrict__ FB)
{
    int s = blockIdx.x;
    for (int i = threadIdx.x; i < S_DIM; i += 256) {
        int idx = s * S_DIM + i;
        D[idx]  = -G[idx];
        YB[idx] = Y[idx];
    }
    if (threadIdx.x == 0) FB[s] = FC[s];
}

// f32 -> f16 conversion (weights, once per call)
__global__ void k_cvt16(const float* __restrict__ src,
                        _Float16* __restrict__ dst, int n)
{
    int i = blockIdx.x * blockDim.x + threadIdx.x;
    if (i < n) dst[i] = (_Float16)src[i];
}

// copy f32 -> f32 + f16 mirror
__global__ void k_copy_dual(const float* __restrict__ src,
                            float* __restrict__ dst,
                            _Float16* __restrict__ dst16, int n)
{
    int i = blockIdx.x * blockDim.x + threadIdx.x;
    if (i < n) { float v = src[i]; dst[i] = v; dst16[i] = (_Float16)v; }
}

__global__ void k_copy(const float* __restrict__ src, float* __restrict__ dst, int n)
{
    int i = blockIdx.x * blockDim.x + threadIdx.x;
    if (i < n) dst[i] = src[i];
}

// ---------------------------------------------------------------------------
// Host orchestration
// ---------------------------------------------------------------------------
extern "C" void kernel_launch(void* const* d_in, const int* in_sizes, int n_in,
                              void* d_out, int out_size, void* d_ws, size_t ws_size,
                              hipStream_t stream)
{
    (void)in_sizes; (void)n_in; (void)out_size; (void)ws_size;
    const float* W0 = (const float*)d_in[0];   // (H,S)
    const float* b0 = (const float*)d_in[1];
    const float* W1 = (const float*)d_in[2];   // (H,H)
    const float* b1 = (const float*)d_in[3];
    const float* W2 = (const float*)d_in[4];   // (O,H)
    const float* b2 = (const float*)d_in[5];
    const float* Y0 = (const float*)d_in[6];   // (B,S)
    const float* TO = (const float*)d_in[7];   // (B,O)
    const int*   MK = (const int*)d_in[8];     // (B,O)

    // ---- workspace layout: f32 region, then f16 region --------------------
    float* p   = (float*)d_ws;
    float* H1  = p; p += BATCH * H_DIM;
    float* H2  = p; p += BATCH * H_DIM;
    float* OB  = p; p += BATCH * O_DIM;
    float* DO_ = p; p += BATCH * O_DIM;
    float* DH2 = p; p += BATCH * H_DIM;
    float* DH1 = p; p += BATCH * H_DIM;
    float* Y   = p; p += BATCH * S_DIM;
    float* YN  = p; p += BATCH * S_DIM;
    float* G   = p; p += BATCH * S_DIM;
    float* GN  = p; p += BATCH * S_DIM;
    float* D   = p; p += BATCH * S_DIM;
    float* FC  = p; p += BATCH;
    float* FB  = p; p += BATCH;
    float* FT  = p; p += BATCH;
    float* FN  = p; p += BATCH;
    float* GD  = p; p += BATCH;
    _Float16* q    = (_Float16*)p;
    _Float16* W0h  = q; q += H_DIM * S_DIM;
    _Float16* W1h  = q; q += H_DIM * H_DIM;
    _Float16* W2h  = q; q += O_DIM * H_DIM;
    _Float16* H1h  = q; q += BATCH * H_DIM;
    _Float16* H2h  = q; q += BATCH * H_DIM;
    _Float16* DO16 = q; q += BATCH * O_DIM;
    _Float16* DH2h = q; q += BATCH * H_DIM;
    _Float16* DH1h = q; q += BATCH * H_DIM;
    _Float16* Yh   = q; q += BATCH * S_DIM;
    _Float16* YTh  = q; q += BATCH * S_DIM;
    _Float16* YNh  = q; q += BATCH * S_DIM;
    float* YB = (float*)d_out;                 // y_best lives in d_out

    auto gemm = [&](const _Float16* A, const _Float16* W, int wtrans,
                    const float* bias, const float* mask,
                    float* C, _Float16* C16, int M, int N, int K, int relu) {
        dim3 grid(N / TN, M / TM);
        k_gemm_wmma<<<grid, GEMM_THREADS, 0, stream>>>(A, W, wtrans, bias, mask,
                                                       C, C16, M, N, K, relu);
    };
    auto forward = [&](const _Float16* yh) {
        gemm(yh,  W0h, 1, b0, nullptr, H1, H1h, BATCH, H_DIM, S_DIM, 1);
        gemm(H1h, W1h, 1, b1, nullptr, H2, H2h, BATCH, H_DIM, H_DIM, 1);
        gemm(H2h, W2h, 1, b2, nullptr, OB, nullptr, BATCH, O_DIM, H_DIM, 0);
    };
    auto backward = [&]() {
        gemm(DO16, W2h, 0, nullptr, H2, DH2, DH2h, BATCH, H_DIM, O_DIM, 0);
        gemm(DH2h, W1h, 0, nullptr, H1, DH1, DH1h, BATCH, H_DIM, H_DIM, 0);
        gemm(DH1h, W0h, 0, nullptr, nullptr, GN, nullptr, BATCH, S_DIM, H_DIM, 0);
    };

    // ---- weights -> f16 once per call -------------------------------------
    k_cvt16<<<(H_DIM * S_DIM + 255) / 256, 256, 0, stream>>>(W0, W0h, H_DIM * S_DIM);
    k_cvt16<<<(H_DIM * H_DIM + 255) / 256, 256, 0, stream>>>(W1, W1h, H_DIM * H_DIM);
    k_cvt16<<<(O_DIM * H_DIM + 255) / 256, 256, 0, stream>>>(W2, W2h, O_DIM * H_DIM);

    // ---- init: y = y0; f0,g0 = value_and_grad(y0); d=-g0; best = y0,f0 ----
    {
        int n = BATCH * S_DIM;
        k_copy_dual<<<(n + 255) / 256, 256, 0, stream>>>(Y0, Y, Yh, n);
    }
    forward(Yh);
    k_loss<<<BATCH, 256, 0, stream>>>(OB, TO, MK, FC, DO_, DO16);
    backward();
    {
        int n = BATCH * S_DIM;
        k_copy<<<(n + 255) / 256, 256, 0, stream>>>(GN, G, n);
    }
    k_init<<<BATCH, 256, 0, stream>>>(G, D, Y, YB, FC, FB);

    // ---- 24 CG steps ------------------------------------------------------
    for (int it = 0; it < NSTEPS; ++it) {
        k_dirfix_trial<<<BATCH, 256, 0, stream>>>(G, D, Y, YTh, GD);
        forward(YTh);
        k_loss<<<BATCH, 256, 0, stream>>>(OB, TO, MK, FT, nullptr, nullptr);
        k_alpha_step<<<BATCH, 256, 0, stream>>>(Y, D, FT, FC, GD, YN, YNh);
        forward(YNh);
        k_loss<<<BATCH, 256, 0, stream>>>(OB, TO, MK, FN, DO_, DO16);
        backward();
        k_finalize<<<BATCH, 256, 0, stream>>>(GN, G, D, YN, Y, FN, FC, YB, FB);
    }
    // d_out already holds y_best
}
